// FSEncoder_SEAN_44779329028472
// MI455X (gfx1250) — matrix operand order repaired and checked by reading, not compiled, not run on hardware
//
#include <hip/hip_runtime.h>
#include <hip/hip_bf16.h>
#include <math.h>

typedef __attribute__((ext_vector_type(16))) _Float16 v16h;
typedef __attribute__((ext_vector_type(8)))  _Float16 v8h;
typedef __attribute__((ext_vector_type(8)))  float    v8f;

#define IN_EPS 1e-5f

// ---------------------------------------------------------------------------
// f32 -> f16 convert (input image pre-pass)
// ---------------------------------------------------------------------------
__global__ void f2h_kernel(const float* __restrict__ x, _Float16* __restrict__ y, int n) {
    int i = blockIdx.x * blockDim.x + threadIdx.x;
    if (i < n) y[i] = (_Float16)x[i];
}

// ---------------------------------------------------------------------------
// Weight transform: (Cout,Cin,3,3) f32 -> [Cout][Kpad] f16.
// cinShift>0: tap-major K order, k = t*Cin + ci  (Cin = 1<<cinShift)
// cinShift==0: flattened OIHW order, k = ci*9 + t (conv0 only)
// ---------------------------------------------------------------------------
__global__ void wt_kernel(const float* __restrict__ w, _Float16* __restrict__ wo,
                          int Cout, int Cin, int Kpad, int cinShift) {
    int idx = blockIdx.x * blockDim.x + threadIdx.x;
    int total = Cout * Kpad;
    if (idx >= total) return;
    int c = idx / Kpad, kp = idx - c * Kpad;
    int Ktot = Cin * 9;
    float v = 0.f;
    if (kp < Ktot) {
        int src = kp;
        if (cinShift > 0) {
            int t = kp >> cinShift;
            int ci = kp & (Cin - 1);
            src = ci * 9 + t;
        }
        v = w[(size_t)c * Ktot + src];
    }
    wo[idx] = (_Float16)v;
}

// ---------------------------------------------------------------------------
// Implicit-GEMM 3x3 conv via V_WMMA_F32_16X16X32_F16, f16 activations in.
//   Block: 256 threads = 8 waves. Block tile: BLOCK_M pixels x 128 channels.
//   NT = N-tiles per wave (4 -> BLOCK_M=64, 8 -> BLOCK_M=128).
//   Wave: one 16-pixel M-tile x NT 16-channel N-tiles -> NT WMMA / K-chunk,
//   one shared A fragment (2x ds_load_b128). LDS A tile double-buffered.
// TAPMAJOR=1: K-chunk lies in one (kh,kw) tap (Cin pow2, %32==0) -> im2col
//   address math hoisted out of the per-element staging loop.
// PADMODE: 0 = zero pad 1 (strided convs), 1 = reflect pad 1 (stride 1).
// ACT: 0 = none, 1 = tanh (style head).
// Preconditions (hold for every layer here): HWo % BLOCK_M == 0, Wout = 2^woutShift.
// ---------------------------------------------------------------------------
template<int PADMODE, int ACT, int TAPMAJOR, int NT>
__global__ __launch_bounds__(256)
void conv_wmma(const _Float16* __restrict__ x, const _Float16* __restrict__ wq,
               const float* __restrict__ bias, float* __restrict__ y,
               int Cin, int cinShift, int Hin, int Win, int Cout,
               int HWo, int woutShift, int stride, int Kpad) {
    constexpr int WCOL    = 128 / (16 * NT);      // wave columns: 2 (NT=4) or 1 (NT=8)
    constexpr int BLOCK_M = (8 / WCOL) * 16;      // 64 or 128 pixels per block
    constexpr int G       = BLOCK_M / 64;         // staging pixel groups per thread

    const int b      = blockIdx.z;
    const size_t HWi = (size_t)Hin * Win;
    const int m_base = blockIdx.x * BLOCK_M;
    const int n_base = blockIdx.y * 128;
    const int Ktot   = Cin * 9;
    const int wmask  = (1 << woutShift) - 1;

    __shared__ __align__(16) _Float16 sA[2 * BLOCK_M * 40];

    const int tid  = threadIdx.x;
    const int lane = tid & 31;
    const int wv   = tid >> 5;            // 0..7
    const int mt   = wv / WCOL;           // M tile of this wave
    const int ncg  = wv % WCOL;           // N group of this wave
    const int hi   = lane >> 4;
    const int ln   = lane & 15;
    const int row  = mt * 16 + ln;

    v8f acc[NT];
    #pragma unroll
    for (int j = 0; j < NT; ++j) acc[j] = (v8f){0.f,0.f,0.f,0.f,0.f,0.f,0.f,0.f};

    // ---- staging: thread owns G pixels (sp, sp+64) x 8 consecutive K slots ----
    const int sp = tid & 63;
    const int ks = (tid >> 6) * 8;        // 0,8,16,24
    int oh[G], ow[G];
    #pragma unroll
    for (int g = 0; g < G; ++g) {
        int p = m_base + sp + g * 64;     // always < HWo (exact grid)
        oh[g] = p >> woutShift;
        ow[g] = p & wmask;
    }
    const _Float16* xb = x + (size_t)b * Cin * HWi;

    // ---- weight pointers for this wave's NT N-tiles ----
    const _Float16* wb[NT];
    int cch[NT];
    #pragma unroll
    for (int j = 0; j < NT; ++j) {
        cch[j] = n_base + ncg * (NT * 16) + j * 16 + ln;
        wb[j] = (cch[j] < Cout) ? (wq + (size_t)cch[j] * Kpad) : (const _Float16*)0;
    }

    int cidx = 0;
    for (int k0 = 0; k0 < Ktot; k0 += 32, ++cidx) {
        _Float16* sbuf = sA + (cidx & 1) * (BLOCK_M * 40);

        // ------------------ stage A tile ------------------
        if (TAPMAJOR) {
            int t  = k0 >> cinShift;      // tap, uniform across chunk
            int kh = t / 3, kw2 = t - kh * 3;
            int cb = (k0 & (Cin - 1)) + ks;
            const _Float16* xc = xb + (size_t)cb * HWi;
            #pragma unroll
            for (int g = 0; g < G; ++g) {
                int ih = oh[g] * stride + kh - 1;
                int iw = ow[g] * stride + kw2 - 1;
                bool inb = true;
                if (PADMODE == 1) {       // reflect: always in-bounds
                    ih = (ih < 0) ? -ih : ((ih >= Hin) ? 2 * Hin - 2 - ih : ih);
                    iw = (iw < 0) ? -iw : ((iw >= Win) ? 2 * Win - 2 - iw : iw);
                } else {
                    inb = (ih >= 0) && (ih < Hin) && (iw >= 0) && (iw < Win);
                }
                const _Float16* xp = xc + (inb ? ((size_t)ih * Win + iw) : (size_t)0);
                v8h sv;
                #pragma unroll
                for (int e = 0; e < 8; ++e)
                    sv[e] = inb ? xp[(size_t)e * HWi] : (_Float16)0.f;
                *(v8h*)(sbuf + (sp + g * 64) * 40 + ks) = sv;   // ds_store_b128
            }
        } else {                          // conv0 fallback (Cin=3, K=27, one chunk)
            #pragma unroll
            for (int g = 0; g < G; ++g) {
                v8h sv;
                #pragma unroll
                for (int e = 0; e < 8; ++e) {
                    int kg = k0 + ks + e;
                    _Float16 v = (_Float16)0.f;
                    if (kg < Ktot) {
                        int ci = kg / 9, r = kg - ci * 9;
                        int kh = r / 3, kw2 = r - kh * 3;
                        int ih = oh[g] * stride + kh - 1;
                        int iw = ow[g] * stride + kw2 - 1;
                        if (PADMODE == 1) {
                            ih = (ih < 0) ? -ih : ((ih >= Hin) ? 2 * Hin - 2 - ih : ih);
                            iw = (iw < 0) ? -iw : ((iw >= Win) ? 2 * Win - 2 - iw : iw);
                            v = xb[(size_t)ci * HWi + (size_t)ih * Win + iw];
                        } else if (ih >= 0 && ih < Hin && iw >= 0 && iw < Win) {
                            v = xb[(size_t)ci * HWi + (size_t)ih * Win + iw];
                        }
                    }
                    sv[e] = v;
                }
                *(v8h*)(sbuf + (sp + g * 64) * 40 + ks) = sv;
            }
        }
        __syncthreads();

        // ------------------ A fragment (2x ds_load_b128) ------------------
        v8h a0 = *(const v8h*)(sbuf + row * 40 + 8 * hi);
        v8h a1 = *(const v8h*)(sbuf + row * 40 + 16 + 8 * hi);
        v16h af;
        #pragma unroll
        for (int j = 0; j < 8; ++j) { af[j] = a0[j]; af[j + 8] = a1[j]; }

        // ------------------ NT independent WMMAs, shared A ------------------
        #pragma unroll
        for (int j = 0; j < NT; ++j) {
            v16h bf;
            if (wb[j]) {
                const _Float16* wp = wb[j] + k0 + 8 * hi;
                v8h b0 = *(const v8h*)(wp);
                v8h b1 = *(const v8h*)(wp + 16);
                #pragma unroll
                for (int e = 0; e < 8; ++e) { bf[e] = b0[e]; bf[e + 8] = b1[e]; }
            } else {
                #pragma unroll
                for (int e = 0; e < 16; ++e) bf[e] = (_Float16)0.f;
            }
            acc[j] = __builtin_amdgcn_wmma_f32_16x16x32_f16(false, af, false, bf,
                                                            (short)0, acc[j], false, false);
        }
        if (wb[0]) __builtin_prefetch(wb[0] + k0 + 32, 0, 3);   // global_prefetch_b8
        // double buffer: single barrier per chunk
    }

    // ------------------ epilogue: bias (+tanh), store D tiles ------------------
    const int pbase = m_base + mt * 16 + 8 * hi;
    #pragma unroll
    for (int j = 0; j < NT; ++j) {
        if (cch[j] < Cout) {
            float bv = bias[cch[j]];
            float* yb = y + ((size_t)b * Cout + cch[j]) * HWo + pbase;
            #pragma unroll
            for (int r = 0; r < 8; ++r) {
                float v = acc[j][r] + bv;
                if (ACT == 1) v = tanhf(v);
                yb[r] = v;
            }
        }
    }
}

// ---------------------------------------------------------------------------
// InstanceNorm2d(affine=False) + LeakyReLU(0.2). One block per (n,c).
// Reads f32 conv output, writes T (f16 activations, f32 for final output).
// ---------------------------------------------------------------------------
template<typename T>
__global__ __launch_bounds__(1024)
void inorm_lrelu_kernel(const float* __restrict__ x, T* __restrict__ y, int HW) {
    const size_t base = (size_t)blockIdx.x * HW;
    const float* xp = x + base;
    T* yp = y + base;
    float s = 0.f, ss = 0.f;
    for (int i = threadIdx.x; i < HW; i += 1024) {
        float v = xp[i]; s += v; ss += v * v;
    }
    __shared__ float rs[1024], rq[1024];
    rs[threadIdx.x] = s; rq[threadIdx.x] = ss;
    __syncthreads();
    for (int off = 512; off > 0; off >>= 1) {
        if (threadIdx.x < off) {
            rs[threadIdx.x] += rs[threadIdx.x + off];
            rq[threadIdx.x] += rq[threadIdx.x + off];
        }
        __syncthreads();
    }
    float mean = rs[0] / (float)HW;
    float var  = rq[0] / (float)HW - mean * mean;
    float inv  = rsqrtf(var + IN_EPS);
    for (int i = threadIdx.x; i < HW; i += 1024) {
        float v = (xp[i] - mean) * inv;
        v = (v >= 0.f) ? v : 0.2f * v;
        yp[i] = (T)v;
    }
}

// ---------------------------------------------------------------------------
// Nearest-resize one-hot segmap -> int label map. src = floor(dst * H/h).
// ---------------------------------------------------------------------------
__global__ void labmap_kernel(const float* __restrict__ seg, int* __restrict__ lab,
                              int B, int S, int H, int W, int h, int w) {
    int idx = blockIdx.x * blockDim.x + threadIdx.x;
    int total = B * h * w;
    if (idx >= total) return;
    int b = idx / (h * w), p = idx - b * (h * w);
    int yy = p / w, xx = p - yy * w;
    int sy = (int)((long long)yy * H / h);
    int sx = (int)((long long)xx * W / w);
    int l = -1;
    for (int s = 0; s < S; ++s) {
        float v = seg[(((size_t)b * S + s) * H + sy) * W + sx];
        if (v != 0.f) { l = s; break; }
    }
    lab[idx] = l;
}

__global__ __launch_bounds__(256)
void counts_kernel(const int* __restrict__ lab, float* __restrict__ counts,
                   int S, int HW) {
    int b = blockIdx.x / S, s = blockIdx.x - b * S;
    const int* lp = lab + (size_t)b * HW;
    int c = 0;
    for (int i = threadIdx.x; i < HW; i += 256) c += (lp[i] == s) ? 1 : 0;
    __shared__ int r[256];
    r[threadIdx.x] = c; __syncthreads();
    for (int off = 128; off > 0; off >>= 1) {
        if (threadIdx.x < off) r[threadIdx.x] += r[threadIdx.x + off];
        __syncthreads();
    }
    if (threadIdx.x == 0) counts[blockIdx.x] = (float)r[0];
}

// ---------------------------------------------------------------------------
// Fused scatter-mean: block per (b,f); 12 per-segment partials in registers.
// ---------------------------------------------------------------------------
__global__ __launch_bounds__(256)
void segmean_kernel(const float* __restrict__ codes, const int* __restrict__ lab,
                    const float* __restrict__ counts, float* __restrict__ out,
                    int S, int F, int HW) {
    int b = blockIdx.x / F, f = blockIdx.x - b * F;
    const float* cp = codes + ((size_t)b * F + f) * HW;
    const int* lp = lab + (size_t)b * HW;
    float acc[12];
    #pragma unroll
    for (int s = 0; s < 12; ++s) acc[s] = 0.f;
    for (int i = threadIdx.x; i < HW; i += 256) {
        int l = lp[i];
        float v = cp[i];
        #pragma unroll
        for (int s = 0; s < 12; ++s) acc[s] += (l == s) ? v : 0.f;
    }
    __shared__ float r[12][256];
    #pragma unroll
    for (int s = 0; s < 12; ++s) r[s][threadIdx.x] = acc[s];
    __syncthreads();
    for (int off = 128; off > 0; off >>= 1) {
        if (threadIdx.x < off) {
            #pragma unroll
            for (int s = 0; s < 12; ++s) r[s][threadIdx.x] += r[s][threadIdx.x + off];
        }
        __syncthreads();
    }
    if (threadIdx.x < S) {
        int s = threadIdx.x;
        float cnt = counts[b * S + s];
        float v = (cnt > 0.f) ? (r[s][0] / fmaxf(cnt, 1.f)) : 0.f;
        out[((size_t)b * S + s) * F + f] = v;
    }
}

// ---------------------------------------------------------------------------
extern "C" void kernel_launch(void* const* d_in, const int* in_sizes, int n_in,
                              void* d_out, int out_size, void* d_ws, size_t ws_size,
                              hipStream_t stream) {
    const int B = 2, S = 12, H = 1024, W = 1024;

    const float* x   = (const float*)d_in[0];
    const float* seg = (const float*)d_in[1];
    const float* w0  = (const float*)d_in[2];  const float* b0  = (const float*)d_in[3];
    const float* w1  = (const float*)d_in[4];  const float* b1  = (const float*)d_in[5];
    const float* w2  = (const float*)d_in[6];  const float* b2  = (const float*)d_in[7];
    const float* w3  = (const float*)d_in[8];  const float* b3  = (const float*)d_in[9];
    const float* wsw = (const float*)d_in[10]; const float* bsb = (const float*)d_in[11];
    const float* s1w = (const float*)d_in[12]; const float* s1b = (const float*)d_in[13];
    const float* s2w = (const float*)d_in[14]; const float* s2b = (const float*)d_in[15];
    const float* s3w = (const float*)d_in[16]; const float* s3b = (const float*)d_in[17];

    // ---- workspace arena (regions reused along the pipeline; peak ~500 MB) ----
    char* base = (char*)d_ws;
    size_t off = 0;
    auto take = [&](size_t bytes) -> char* {
        char* r = base + off; off += (bytes + 255) & ~(size_t)255; return r;
    };
    char* YAR = take(268435456);            // f32 conv outputs (reused every layer)
    char* XA  = take(134217728);            // f16 activations (even layers)
    char* XB  = take(67108864);             // f16 activations (odd layers)
    _Float16* XIN = (_Float16*)take((size_t)B * 3 * H * W * 2);
    int*   LAB = (int*)take((size_t)B * 128 * 128 * 4);
    float* CNT = (float*)take(B * S * 4);
    _Float16* w0t = (_Float16*)take((size_t)32  * 32   * 2);
    _Float16* w1t = (_Float16*)take((size_t)64  * 288  * 2);
    _Float16* w2t = (_Float16*)take((size_t)128 * 576  * 2);
    _Float16* w3t = (_Float16*)take((size_t)256 * 1152 * 2);
    _Float16* wst = (_Float16*)take((size_t)512 * 2304 * 2);
    _Float16* s1t = (_Float16*)take((size_t)512 * 2304 * 2);
    _Float16* s2t = (_Float16*)take((size_t)512 * 4608 * 2);
    _Float16* s3t = (_Float16*)take((size_t)512 * 4608 * 2);

    float* Y      = (float*)YAR;            // every conv's f32 output
    float* CODES  = (float*)YAR;            // style output (Y3 dead by then)
    float* YS     = (float*)(YAR + 134217728);  // structure-head conv outputs
    _Float16* X0 = (_Float16*)XA;           // 1024^2 x32
    _Float16* X1 = (_Float16*)XB;           // 512^2  x64
    _Float16* X2 = (_Float16*)XA;           // 256^2  x128
    _Float16* X3 = (_Float16*)XB;           // 128^2  x256 (feats, lives long)
    _Float16* X4 = (_Float16*)XA;           // 64^2   x512
    _Float16* X5 = (_Float16*)XB;           // 32^2   x512 (X3 dead by then)

    // ---- input -> f16, weights -> tap-major f16 ----
    {
        int n = B * 3 * H * W;
        f2h_kernel<<<(n + 255) / 256, 256, 0, stream>>>(x, XIN, n);
    }
    auto wt = [&](const float* w, _Float16* o, int Cout, int Cin, int Kpad, int shf) {
        int total = Cout * Kpad;
        wt_kernel<<<(total + 255) / 256, 256, 0, stream>>>(w, o, Cout, Cin, Kpad, shf);
    };
    wt(w0,  w0t, 32,  3,   32,   0);
    wt(w1,  w1t, 64,  32,  288,  5);
    wt(w2,  w2t, 128, 64,  576,  6);
    wt(w3,  w3t, 256, 128, 1152, 7);
    wt(wsw, wst, 512, 256, 2304, 8);
    wt(s1w, s1t, 512, 256, 2304, 8);
    wt(s2w, s2t, 512, 512, 4608, 9);
    wt(s3w, s3t, 512, 512, 4608, 9);

    auto cgrid = [&](int HWo, int Cout, int bm) {
        return dim3(HWo / bm, (Cout + 127) / 128, B);
    };

    // ---- backbone ----
    conv_wmma<1, 0, 0, 4><<<cgrid(1024 * 1024, 32, 64), 256, 0, stream>>>(
        XIN, w0t, b0, Y, 3, 0, 1024, 1024, 32, 1024 * 1024, 10, 1, 32);
    inorm_lrelu_kernel<_Float16><<<B * 32, 1024, 0, stream>>>(Y, X0, 1024 * 1024);

    conv_wmma<0, 0, 1, 4><<<cgrid(512 * 512, 64, 64), 256, 0, stream>>>(
        X0, w1t, b1, Y, 32, 5, 1024, 1024, 64, 512 * 512, 9, 2, 288);
    inorm_lrelu_kernel<_Float16><<<B * 64, 1024, 0, stream>>>(Y, X1, 512 * 512);

    conv_wmma<0, 0, 1, 8><<<cgrid(256 * 256, 128, 128), 256, 0, stream>>>(
        X1, w2t, b2, Y, 64, 6, 512, 512, 128, 256 * 256, 8, 2, 576);
    inorm_lrelu_kernel<_Float16><<<B * 128, 1024, 0, stream>>>(Y, X2, 256 * 256);

    conv_wmma<0, 0, 1, 8><<<cgrid(128 * 128, 256, 128), 256, 0, stream>>>(
        X2, w3t, b3, Y, 128, 7, 256, 256, 256, 128 * 128, 7, 2, 1152);
    inorm_lrelu_kernel<_Float16><<<B * 256, 1024, 0, stream>>>(Y, X3, 128 * 128);

    // ---- style head (tanh fused) + segment scatter-mean ----
    conv_wmma<1, 1, 1, 8><<<cgrid(128 * 128, 512, 128), 256, 0, stream>>>(
        X3, wst, bsb, CODES, 256, 8, 128, 128, 512, 128 * 128, 7, 1, 2304);
    labmap_kernel<<<(B * 128 * 128 + 255) / 256, 256, 0, stream>>>(seg, LAB, B, S, H, W, 128, 128);
    counts_kernel<<<B * S, 256, 0, stream>>>(LAB, CNT, S, 128 * 128);
    segmean_kernel<<<B * 512, 256, 0, stream>>>(CODES, LAB, CNT, (float*)d_out, S, 512, 128 * 128);

    // ---- structure head ----
    conv_wmma<0, 0, 1, 8><<<cgrid(64 * 64, 512, 128), 256, 0, stream>>>(
        X3, s1t, s1b, YS, 256, 8, 128, 128, 512, 64 * 64, 6, 2, 2304);
    inorm_lrelu_kernel<_Float16><<<B * 512, 1024, 0, stream>>>(YS, X4, 64 * 64);

    conv_wmma<0, 0, 1, 8><<<cgrid(32 * 32, 512, 128), 256, 0, stream>>>(
        X4, s2t, s2b, YS, 512, 9, 64, 64, 512, 32 * 32, 5, 2, 4608);
    inorm_lrelu_kernel<_Float16><<<B * 512, 1024, 0, stream>>>(YS, X5, 32 * 32);

    conv_wmma<0, 0, 1, 8><<<cgrid(16 * 16, 512, 128), 256, 0, stream>>>(
        X5, s3t, s3b, YS, 512, 9, 32, 32, 512, 16 * 16, 4, 2, 4608);
    inorm_lrelu_kernel<float><<<B * 512, 1024, 0, stream>>>(
        YS, (float*)d_out + (size_t)B * S * 512, 16 * 16);
}
